// VQAttention_13271448944642
// MI455X (gfx1250) — compile-verified, matrix operand order checked by reading, not run on hardware
//
#include <hip/hip_runtime.h>
#include <hip/hip_bf16.h>
#include <math.h>

// Problem constants (match reference)
#define B_  8
#define H_  8
#define L_  512
#define M_  512
#define S_  512
#define DK_ 128
#define DV_ 128
#define D_  1024
#define W_  1024   // M + L
#define CT_ 1536   // S + W  (concatenated score width)
#define SC_Q 0.29730177875068026f  // 128^(-1/4) = TAU^(-1/2)

typedef __attribute__((ext_vector_type(16))) __bf16        v16bf;
typedef __attribute__((ext_vector_type(8)))  float         v8f;
typedef __attribute__((ext_vector_type(4)))  unsigned int  u32x4;

union FragU { v16bf v; u32x4 q[2]; };

__device__ __forceinline__ unsigned short f2bf(float f) {
  unsigned int u = __float_as_uint(f);
  unsigned int r = (u + 0x7FFFu + ((u >> 16) & 1u)) >> 16;  // RNE
  return (unsigned short)r;
}

// Load one lane's 16-element bf16 fragment: elems[0..7] at p, elems[8..15] at p+16
// (K-octets per the CDNA5 16-bit A/B 16x32 layout; caller adds half*8 to p).
__device__ __forceinline__ v16bf load_frag16(const unsigned short* p) {
  FragU f;
  f.q[0] = *(const u32x4*)(p);
  f.q[1] = *(const u32x4*)(p + 16);
  return f.v;
}

//==========================================================================
// LayerNorm of x (row = b*L + l, D=1024) -> bf16
//==========================================================================
__global__ void k_ln_x(const float* __restrict__ x, const float* __restrict__ g,
                       const float* __restrict__ bb, unsigned short* __restrict__ xt) {
  __shared__ float red[256];
  int row = blockIdx.x;
  int tid = threadIdx.x;
  const float* xr = x + (size_t)row * D_;
  float v[4]; float s = 0.f;
  for (int t = 0; t < 4; ++t) { v[t] = xr[tid + t * 256]; s += v[t]; }
  red[tid] = s; __syncthreads();
  for (int o = 128; o > 0; o >>= 1) { if (tid < o) red[tid] += red[tid + o]; __syncthreads(); }
  float mu = red[0] * (1.f / D_); __syncthreads();
  s = 0.f;
  for (int t = 0; t < 4; ++t) { float d = v[t] - mu; s += d * d; }
  red[tid] = s; __syncthreads();
  for (int o = 128; o > 0; o >>= 1) { if (tid < o) red[tid] += red[tid + o]; __syncthreads(); }
  float rstd = rsqrtf(red[0] * (1.f / D_) + 1e-6f);
  unsigned short* out = xt + (size_t)row * D_;
  for (int t = 0; t < 4; ++t) {
    int c = tid + t * 256;
    out[c] = f2bf((v[t] - mu) * rstd * g[c] + bb[c]);
  }
}

//==========================================================================
// Transpose + convert weight: src f32 (K x N) -> dst bf16 (N x K)
//==========================================================================
__global__ void k_w2bf_t(const float* __restrict__ src, unsigned short* __restrict__ dst,
                         int K, int N) {
  size_t i = (size_t)blockIdx.x * blockDim.x + threadIdx.x;
  if (i >= (size_t)K * N) return;
  int k = (int)(i / N), n = (int)(i % N);
  dst[(size_t)n * K + k] = f2bf(src[i]);
}

//==========================================================================
// Generic batched WMMA GEMM: C[M,N] (+)= A[M,K] * Bt[N,K]^T  (bf16 in, f32 out)
// mode 0: store.  mode 1: rel-shift scatter-add (bd path, N=1024, ccol=512).
//==========================================================================
__global__ void k_gemm_bf16(const unsigned short* __restrict__ A, int lda, long long sA,
                            const unsigned short* __restrict__ Bt, int ldb, long long sB, int bmod,
                            float* __restrict__ C, int ldc, long long sC, int ccol,
                            int Mr, int Nc, int K, int mode) {
  int bz = blockIdx.z;
  A  += (size_t)bz * (size_t)sA;
  Bt += (size_t)(bmod ? (bz % bmod) : bz) * (size_t)sB;
  C  += (size_t)bz * (size_t)sC;
  int ntile = Nc >> 4;
  int w  = blockIdx.x * (blockDim.x >> 5) + (threadIdx.x >> 5);
  int tm = w / ntile, tn = w - tm * ntile;
  if (tm >= (Mr >> 4)) return;            // wave-uniform
  if (mode == 1 && tn < 31) return;       // bd: columns < 511 never used (locality mask)
  int lane = threadIdx.x & 31;
  int hf = lane >> 4, r = lane & 15;
  const unsigned short* arow = A  + (size_t)(tm * 16 + r) * lda + hf * 8;
  const unsigned short* brow = Bt + (size_t)(tn * 16 + r) * ldb + hf * 8;
  v8f acc = {};
  for (int k0 = 0; k0 < K; k0 += 32) {
    __builtin_prefetch(arow + k0 + 64, 0, 1);
    __builtin_prefetch(brow + k0 + 64, 0, 1);
    v16bf a = load_frag16(arow + k0);
    v16bf b = load_frag16(brow + k0);
    acc = __builtin_amdgcn_wmma_f32_16x16x32_bf16(false, a, false, b, (short)0, acc, false, false);
  }
  if (mode == 0) {
    float* cc = C + ccol + tn * 16 + r;
    for (int e = 0; e < 8; ++e) {
      int m = tm * 16 + (e + 8 * hf);
      cc[(size_t)m * ldc] = acc[e];
    }
  } else {
    int n = tn * 16 + r;
    if (n >= 511) {
      for (int e = 0; e < 8; ++e) {
        int m = tm * 16 + (e + 8 * hf);
        int j = n + m - 511;               // rel_shift: bd[m, j] = P[m, n]
        if (j < W_) C[(size_t)m * ldc + S_ + j] += acc[e];
      }
    }
  }
}

//==========================================================================
// Codebook prep: bf16 copy + row norms
//==========================================================================
__global__ void k_cb_prep(const float* __restrict__ cbf, unsigned short* __restrict__ cb,
                          float* __restrict__ cnorm) {
  __shared__ float red[128];
  int row = blockIdx.x;   // h*S + s
  int d = threadIdx.x;
  float v = cbf[(size_t)row * DK_ + d];
  cb[(size_t)row * DK_ + d] = f2bf(v);
  red[d] = v * v; __syncthreads();
  for (int o = 64; o > 0; o >>= 1) { if (d < o) red[d] += red[d + o]; __syncthreads(); }
  if (d == 0) cnorm[row] = red[0];
}

//==========================================================================
// Cache tensors -> bf16 buffers (recentK rows [0,M), VT cols: [0,S)=udl^T,
// [S,S+M)=cache_v^T)
//==========================================================================
__global__ void k_cache_prep(const float* __restrict__ ck, const float* __restrict__ cv,
                             const float* __restrict__ udl,
                             unsigned short* __restrict__ recentK,
                             unsigned short* __restrict__ VT) {
  size_t i = (size_t)blockIdx.x * blockDim.x + threadIdx.x;
  if (i >= (size_t)B_ * H_ * M_ * DK_) return;
  int d  = (int)(i & 127);
  int m  = (int)((i >> 7) & 511);
  int bh = (int)(i >> 16);
  recentK[((size_t)bh * W_ + m) * DK_ + d]       = f2bf(ck[i]);
  VT[((size_t)bh * DV_ + d) * CT_ + S_ + m]      = f2bf(cv[i]);
  VT[((size_t)bh * DV_ + d) * CT_ + m]           = f2bf(udl[i]);
}

//==========================================================================
// Per-(b,l,h): LN(q)*sc (+u / +v biases), LN(k)*sc, silu(v) -> transposed VT
//==========================================================================
__global__ void k_qkv_prep(const float* __restrict__ qf, const float* __restrict__ kvg,
                           const float* __restrict__ xl_u, const float* __restrict__ xl_v,
                           unsigned short* __restrict__ QU, unsigned short* __restrict__ QV,
                           unsigned short* __restrict__ KB, unsigned short* __restrict__ VT) {
  __shared__ float red[128];
  int idx = blockIdx.x;        // (b*L + l)*H + h
  int h  = idx & 7;
  int bl = idx >> 3;
  int l  = bl & 511;
  int b  = bl >> 9;
  int d  = threadIdx.x;
  int bh = b * H_ + h;
  size_t qo = ((size_t)bh * L_ + l) * DK_ + d;

  // ---- q: LN over DK, * sc ----
  float q = qf[(size_t)bl * (H_ * DK_) + h * DK_ + d];
  red[d] = q; __syncthreads();
  for (int o = 64; o > 0; o >>= 1) { if (d < o) red[d] += red[d + o]; __syncthreads(); }
  float mu = red[0] * 0.0078125f; __syncthreads();
  float dq = q - mu;
  red[d] = dq * dq; __syncthreads();
  for (int o = 64; o > 0; o >>= 1) { if (d < o) red[d] += red[d + o]; __syncthreads(); }
  float rstd = rsqrtf(red[0] * 0.0078125f + 1e-6f); __syncthreads();
  float qln = dq * rstd * SC_Q;
  QU[qo] = f2bf(qln + xl_u[h * DK_ + d] * SC_Q);
  QV[qo] = f2bf(qln + xl_v[h * DK_ + d] * SC_Q);

  // ---- k: LN over DK, * sc ----
  float k = kvg[(size_t)bl * 3072 + h * DK_ + d];
  red[d] = k; __syncthreads();
  for (int o = 64; o > 0; o >>= 1) { if (d < o) red[d] += red[d + o]; __syncthreads(); }
  float kmu = red[0] * 0.0078125f; __syncthreads();
  float dk = k - kmu;
  red[d] = dk * dk; __syncthreads();
  for (int o = 64; o > 0; o >>= 1) { if (d < o) red[d] += red[d + o]; __syncthreads(); }
  float krstd = rsqrtf(red[0] * 0.0078125f + 1e-6f);
  KB[qo] = f2bf(dk * krstd * SC_Q);

  // ---- v: silu, transposed into VT cols [S+M, CT) ----
  float vv = kvg[(size_t)bl * 3072 + 1024 + h * DV_ + d];
  float sv = vv / (1.f + __expf(-vv));
  VT[((size_t)bh * DV_ + d) * CT_ + S_ + M_ + l] = f2bf(sv);
}

//==========================================================================
// VQ: fused WMMA k.c^T + per-row argmax(dot - ||c||^2/2) + k_hat gather
//==========================================================================
__global__ void k_vq(const unsigned short* __restrict__ kbf,
                     const unsigned short* __restrict__ cb,
                     const float* __restrict__ codebook,
                     const float* __restrict__ cnorm,
                     unsigned short* __restrict__ recentK) {
  int bz = blockIdx.z;                       // b*H + h
  int h  = bz & 7;
  int w  = blockIdx.x * (blockDim.x >> 5) + (threadIdx.x >> 5);  // L/16 tiles
  int lane = threadIdx.x & 31;
  int hf = lane >> 4, r = lane & 15;
  const unsigned short* arow = kbf + ((size_t)bz * L_ + w * 16 + r) * DK_ + hf * 8;
  v16bf a0 = load_frag16(arow + 0);
  v16bf a1 = load_frag16(arow + 32);
  v16bf a2 = load_frag16(arow + 64);
  v16bf a3 = load_frag16(arow + 96);
  const unsigned short* Bt = cb + (size_t)h * S_ * DK_;
  const float* cn = cnorm + h * S_;
  float bestv[8]; int besti[8];
  for (int e = 0; e < 8; ++e) { bestv[e] = -3.0e38f; besti[e] = 0; }
  for (int st = 0; st < (S_ >> 4); ++st) {
    const unsigned short* brow = Bt + (size_t)(st * 16 + r) * DK_ + hf * 8;
    v8f acc = {};
    acc = __builtin_amdgcn_wmma_f32_16x16x32_bf16(false, a0, false, load_frag16(brow + 0),  (short)0, acc, false, false);
    acc = __builtin_amdgcn_wmma_f32_16x16x32_bf16(false, a1, false, load_frag16(brow + 32), (short)0, acc, false, false);
    acc = __builtin_amdgcn_wmma_f32_16x16x32_bf16(false, a2, false, load_frag16(brow + 64), (short)0, acc, false, false);
    acc = __builtin_amdgcn_wmma_f32_16x16x32_bf16(false, a3, false, load_frag16(brow + 96), (short)0, acc, false, false);
    int scol = st * 16 + r;
    float hc = 0.5f * cn[scol];
    for (int e = 0; e < 8; ++e) {
      float val = acc[e] - hc;
      if (val > bestv[e]) { bestv[e] = val; besti[e] = scol; }
      else if (val == bestv[e] && scol < besti[e]) { besti[e] = scol; }
    }
  }
  // butterfly reduce across the 16 lanes that share each C row (wave32, width 16)
  for (int o = 1; o < 16; o <<= 1) {
    for (int e = 0; e < 8; ++e) {
      float ov = __shfl_xor(bestv[e], o, 16);
      int   oi = __shfl_xor(besti[e], o, 16);
      if (ov > bestv[e] || (ov == bestv[e] && oi < besti[e])) { bestv[e] = ov; besti[e] = oi; }
    }
  }
  // gather k_hat rows (16 lanes per half-wave, 8 contiguous elems each)
  for (int e = 0; e < 8; ++e) {
    int l = w * 16 + e + 8 * hf;
    int z = besti[e];
    const float* src = codebook + ((size_t)h * S_ + z) * DK_ + r * 8;
    unsigned short* dst = recentK + ((size_t)bz * W_ + M_ + l) * DK_ + r * 8;
    for (int t = 0; t < 8; ++t) dst[t] = f2bf(src[t]);
  }
}

//==========================================================================
// Sinusoid (flipped) -> bf16 (1024 x 1024)
//==========================================================================
__global__ void k_sin(unsigned short* __restrict__ out) {
  int row = blockIdx.x;
  float p = (float)(1023 - row);
  for (int c = threadIdx.x; c < 1024; c += blockDim.x) {
    int j = c & 511;
    float pre = p * __powf(10000.f, -(float)j * (1.f / 512.f));
    out[(size_t)row * 1024 + c] = f2bf((c < 512) ? __sinf(pre) : __cosf(pre));
  }
}

// r f32 (W x H*DK) -> rt bf16 (H, W, DK) * sc
__global__ void k_rt(const float* __restrict__ rf, unsigned short* __restrict__ rt) {
  size_t i = (size_t)blockIdx.x * blockDim.x + threadIdx.x;
  if (i >= (size_t)H_ * W_ * DK_) return;
  int d = (int)(i & 127);
  int rest = (int)(i >> 7);
  int wpos = rest & 1023;
  int h = rest >> 10;
  rt[i] = f2bf(rf[(size_t)wpos * 1024 + h * DK_ + d] * SC_Q);
}

//==========================================================================
// Fused bias + mask + softmax over concatenated [cache(512) | recent(1024)]
//==========================================================================
__global__ void k_softmax(const float* __restrict__ SCO, const float* __restrict__ agg_low,
                          const int* __restrict__ pos, unsigned short* __restrict__ AA) {
  __shared__ float red[256];
  int row = blockIdx.x;        // (b*H+h)*L + l
  int l  = row & 511;
  int bh = row >> 9;
  int inval = M_ - pos[0]; if (inval < 0) inval = 0;
  const float* sr = SCO + (size_t)row * CT_;
  int tid = threadIdx.x;
  float vals[6];
  float mx = -3.0e38f;
  for (int t = 0; t < 6; ++t) {
    int c = tid + t * 256;
    float v = sr[c];
    if (c < S_) {
      float low = agg_low[(size_t)bh * S_ + c];
      v += (low == 0.f) ? -1e30f : __logf(fmaxf(low, 1.f));
    } else {
      int j = c - S_;
      if (!(j >= inval && j - M_ <= l)) v = -1e30f;
    }
    vals[t] = v;
    mx = fmaxf(mx, v);
  }
  red[tid] = mx; __syncthreads();
  for (int o = 128; o > 0; o >>= 1) { if (tid < o) red[tid] = fmaxf(red[tid], red[tid + o]); __syncthreads(); }
  mx = red[0]; __syncthreads();
  float s = 0.f;
  for (int t = 0; t < 6; ++t) { vals[t] = __expf(vals[t] - mx); s += vals[t]; }
  red[tid] = s; __syncthreads();
  for (int o = 128; o > 0; o >>= 1) { if (tid < o) red[tid] += red[tid + o]; __syncthreads(); }
  float inv = 1.f / red[0];
  unsigned short* ar = AA + (size_t)row * CT_;
  for (int t = 0; t < 6; ++t) ar[tid + t * 256] = f2bf(vals[t] * inv);
}

//==========================================================================
// Gate: gw[b,l,h*DV+d] = silu(g) * wv[b,h,l,d] -> bf16
//==========================================================================
__global__ void k_gate(const float* __restrict__ kvg, const float* __restrict__ wv,
                       unsigned short* __restrict__ gw) {
  size_t i = (size_t)blockIdx.x * blockDim.x + threadIdx.x;
  if (i >= (size_t)B_ * L_ * H_ * DV_) return;
  int hd = (int)(i & 1023);
  int bl = (int)(i >> 10);
  int h = hd >> 7, d = hd & 127;
  int b = bl >> 9, l = bl & 511;
  float g = kvg[(size_t)bl * 3072 + 2048 + hd];
  float gs = g / (1.f + __expf(-g));
  float wvv = wv[(((size_t)(b * H_ + h)) * L_ + l) * DV_ + d];
  gw[i] = f2bf(gs * wvv);
}

//==========================================================================
// Host launcher
//==========================================================================
extern "C" void kernel_launch(void* const* d_in, const int* in_sizes, int n_in,
                              void* d_out, int out_size, void* d_ws, size_t ws_size,
                              hipStream_t stream) {
  const float* x        = (const float*)d_in[0];
  const float* ln_g     = (const float*)d_in[1];
  const float* ln_b     = (const float*)d_in[2];
  const float* Wq       = (const float*)d_in[3];
  const float* Wkvg     = (const float*)d_in[4];
  const float* Wr       = (const float*)d_in[5];
  const float* Wres     = (const float*)d_in[6];
  const float* xl_u     = (const float*)d_in[7];
  const float* xl_v     = (const float*)d_in[8];
  const float* codebook = (const float*)d_in[9];
  const float* cache_k  = (const float*)d_in[10];
  const float* cache_v  = (const float*)d_in[11];
  const float* agg_udl  = (const float*)d_in[12];
  const float* agg_low  = (const float*)d_in[13];
  const int*   pos_off  = (const int*)d_in[14];
  float* out = (float*)d_out;

  char* wsb = (char*)d_ws;
  size_t off = 0;
  auto alloc = [&](size_t bytes) -> void* {
    void* p = wsb + off;
    off = (off + bytes + 255) & ~(size_t)255;
    return p;
  };
  unsigned short* XT    = (unsigned short*)alloc((size_t)4096 * 1024 * 2);     // x_tilde bf16
  unsigned short* WQT   = (unsigned short*)alloc((size_t)1024 * 1024 * 2);
  unsigned short* WKVGT = (unsigned short*)alloc((size_t)3072 * 1024 * 2);
  unsigned short* WRT   = (unsigned short*)alloc((size_t)1024 * 1024 * 2);
  unsigned short* WREST = (unsigned short*)alloc((size_t)1024 * 1024 * 2);
  float*          QF    = (float*)alloc((size_t)4096 * 1024 * 4);              // q raw
  float*          KVGF  = (float*)alloc((size_t)4096 * 3072 * 4);              // kvg raw
  unsigned short* QU    = (unsigned short*)alloc((size_t)64 * 512 * 128 * 2);  // (q+u)*sc
  unsigned short* QV    = (unsigned short*)alloc((size_t)64 * 512 * 128 * 2);  // (q+vb)*sc
  unsigned short* KB    = (unsigned short*)alloc((size_t)64 * 512 * 128 * 2);  // ln(k)*sc
  unsigned short* RECK  = (unsigned short*)alloc((size_t)64 * 1024 * 128 * 2); // [cache_k_hat | k_hat]
  unsigned short* VT    = (unsigned short*)alloc((size_t)64 * 128 * 1536 * 2); // [udl^T | cache_v^T | v^T]
  unsigned short* CB    = (unsigned short*)alloc((size_t)8 * 512 * 128 * 2);
  float*          CN    = (float*)alloc((size_t)8 * 512 * 4);
  unsigned short* SINB  = (unsigned short*)alloc((size_t)1024 * 1024 * 2);
  float*          RF    = (float*)alloc((size_t)1024 * 1024 * 4);
  unsigned short* RT    = (unsigned short*)alloc((size_t)8 * 1024 * 128 * 2);
  float*          SC    = (float*)alloc((size_t)64 * 512 * 1536 * 4);          // concatenated scores
  unsigned short* AA    = (unsigned short*)alloc((size_t)64 * 512 * 1536 * 2); // normalized probs
  float*          WV    = (float*)alloc((size_t)64 * 512 * 128 * 4);
  unsigned short* GW    = (unsigned short*)alloc((size_t)4096 * 1024 * 2);
  (void)in_sizes; (void)n_in; (void)out_size; (void)ws_size;

  // --- stage 1: normalizations / conversions ---
  k_ln_x<<<4096, 256, 0, stream>>>(x, ln_g, ln_b, XT);
  k_w2bf_t<<<(1024 * 1024) / 256, 256, 0, stream>>>(Wq,   WQT,   1024, 1024);
  k_w2bf_t<<<(1024 * 3072) / 256, 256, 0, stream>>>(Wkvg, WKVGT, 1024, 3072);
  k_w2bf_t<<<(1024 * 1024) / 256, 256, 0, stream>>>(Wr,   WRT,   1024, 1024);
  k_w2bf_t<<<(1024 * 1024) / 256, 256, 0, stream>>>(Wres, WREST, 1024, 1024);
  k_cb_prep<<<8 * 512, 128, 0, stream>>>(codebook, CB, CN);
  k_cache_prep<<<(8 * 8 * 512 * 128) / 256, 256, 0, stream>>>(cache_k, cache_v, agg_udl, RECK, VT);
  k_sin<<<1024, 256, 0, stream>>>(SINB);

  // --- stage 2: projection GEMMs ---
  k_gemm_bf16<<<4096, 128, 0, stream>>>(XT, 1024, 0, WQT, 1024, 0, 0,
                                        QF, 1024, 0, 0, 4096, 1024, 1024, 0);
  k_gemm_bf16<<<12288, 128, 0, stream>>>(XT, 1024, 0, WKVGT, 1024, 0, 0,
                                         KVGF, 3072, 0, 0, 4096, 3072, 1024, 0);
  k_gemm_bf16<<<1024, 128, 0, stream>>>(SINB, 1024, 0, WRT, 1024, 0, 0,
                                        RF, 1024, 0, 0, 1024, 1024, 1024, 0);
  k_rt<<<(8 * 1024 * 128) / 256, 256, 0, stream>>>(RF, RT);

  // --- stage 3: per-head tensors + VQ ---
  k_qkv_prep<<<32768, 128, 0, stream>>>(QF, KVGF, xl_u, xl_v, QU, QV, KB, VT);
  k_vq<<<dim3(8, 1, 64), 128, 0, stream>>>(KB, CB, codebook, CN, RECK);

  // --- stage 4: score GEMMs into concatenated buffer ---
  // cache scores: (q+u) . c^T  -> cols [0,512)
  k_gemm_bf16<<<dim3(256, 1, 64), 128, 0, stream>>>(QU, 128, 512LL * 128, CB, 128, 512LL * 128, 8,
                                                    SC, 1536, 512LL * 1536, 0, 512, 512, 128, 0);
  // ac: (q+u) . recent_k_hat^T -> cols [512,1536)
  k_gemm_bf16<<<dim3(512, 1, 64), 128, 0, stream>>>(QU, 128, 512LL * 128, RECK, 128, 1024LL * 128, 0,
                                                    SC, 1536, 512LL * 1536, 512, 512, 1024, 128, 0);
  // bd: (q+vb) . xl_r^T, rel-shift scatter-add into same region
  k_gemm_bf16<<<dim3(512, 1, 64), 128, 0, stream>>>(QV, 128, 512LL * 128, RT, 128, 1024LL * 128, 8,
                                                    SC, 1536, 512LL * 1536, 512, 512, 1024, 128, 1);

  // --- stage 5: mask + softmax (fused), write bf16 probs ---
  k_softmax<<<32768, 256, 0, stream>>>(SC, agg_low, pos_off, AA);

  // --- stage 6: attention . V (single K=1536 GEMM over [udl | cache_v | v]) ---
  k_gemm_bf16<<<dim3(64, 1, 64), 128, 0, stream>>>(AA, 1536, 512LL * 1536, VT, 1536, 128LL * 1536, 0,
                                                   WV, 128, 512LL * 128, 0, 512, 128, 1536, 0);

  // --- stage 7: gate + output projection ---
  k_gate<<<(8 * 512 * 1024) / 256, 256, 0, stream>>>(KVGF, WV, GW);
  k_gemm_bf16<<<4096, 128, 0, stream>>>(GW, 1024, 0, WREST, 1024, 0, 0,
                                        out, 1024, 0, 0, 4096, 1024, 1024, 0);
}
